// MultiHeadedAttention_88295937671520
// MI455X (gfx1250) — matrix operand order repaired and checked by reading, not compiled, not run on hardware
//
#include <hip/hip_runtime.h>
#include <hip/hip_bf16.h>
#include <math.h>

// ---------------------------------------------------------------------------
// MultiHeadedAttention for MI455X (gfx1250).
//   B=2, T=2048, D=1024, H=16, Dk=64.
//   All matmuls via v_wmma_f32_16x16x32_bf16 (f32 accumulate).
//   Attention K/V tiles streamed into LDS by the Tensor Data Mover
//   (tensor_load_to_lds, TENSORcnt double-buffered pipeline).
// ---------------------------------------------------------------------------

typedef __bf16 bf16;
typedef __attribute__((ext_vector_type(8)))  __bf16 v8bf;
typedef __attribute__((ext_vector_type(16))) __bf16 v16bf;
typedef __attribute__((ext_vector_type(8)))  float  v8f;
typedef unsigned int u32x4 __attribute__((ext_vector_type(4)));
typedef int          i32x4 __attribute__((ext_vector_type(4)));
typedef int          i32x8 __attribute__((ext_vector_type(8)));

static constexpr int D_MODEL = 1024;
static constexpr int HEADS   = 16;
static constexpr int DK      = 64;
static constexpr int TSEQ    = 2048;
static constexpr int BATCH   = 2;
static constexpr int NROWS   = BATCH * TSEQ;   // 4096

__device__ __forceinline__ v16bf cat8(v8bf lo, v8bf hi) {
  return __builtin_shufflevector(lo, hi, 0,1,2,3,4,5,6,7,8,9,10,11,12,13,14,15);
}

__device__ __forceinline__ v8f wmma_bf16(v16bf a, v16bf b, v8f c) {
  // D = A(16x32) x B(32x16) + C, f32 accumulate
  return __builtin_amdgcn_wmma_f32_16x16x32_bf16(false, a, false, b,
                                                 (short)0, c, false, false);
}

// ---------------------------------------------------------------------------
// TDM: DMA one 64x64 bf16 2-D tile (row stride `stride_elems`) from global
// memory into LDS at byte offset `lds_off`. Descriptor per CDNA5 ISA §8.3-8.6.
// Tracked with TENSORcnt; EXEC is ignored for tensor ops.
// ---------------------------------------------------------------------------
__device__ __forceinline__ void tdm_load_tile64(
    unsigned lds_off, const bf16* gptr, unsigned long long stride_elems,
    unsigned tdim0, unsigned tdim1) {
  const unsigned long long ga = (unsigned long long)(uintptr_t)gptr;
  // group 0 (128b): count=1 | lds_addr | global_addr[56:0] | type=2
  u32x4 g0 = {1u,
              lds_off,
              (unsigned)ga,
              (unsigned)((ga >> 32) & 0x1FFFFFFu) | (2u << 30)};
  // group 1 (256b): workgroup_mask=0, data_size=1 (2B), tensor dims/strides,
  // tile_dim0=64 (contiguous), tile_dim1=64 rows, tile_dim2=0
  const unsigned tile_w = 64u, tile_h = 64u;
  i32x8 g1 = {(int)(1u << 16),
              (int)((tdim0 & 0xFFFFu) << 16),
              (int)((tdim0 >> 16) | ((tdim1 & 0xFFFFu) << 16)),
              (int)((tdim1 >> 16) | (tile_w << 16)),
              (int)tile_h,
              (int)(unsigned)stride_elems,
              (int)(unsigned)((stride_elems >> 32) & 0xFFFFu),
              0};
  i32x4 gz = {0, 0, 0, 0};
#if defined(__clang_major__) && (__clang_major__ >= 23)
  i32x8 z8 = {0, 0, 0, 0, 0, 0, 0, 0};
  __builtin_amdgcn_tensor_load_to_lds(g0, g1, gz, gz, z8, 0);
#else
  __builtin_amdgcn_tensor_load_to_lds(g0, g1, gz, gz, 0);
#endif
}

// ---------------------------------------------------------------------------
// GEMM: C[N=4096, M=1024] = A[N,1024] @ W[M,1024]^T + bias
// Block: 256 threads (8 waves). Block tile 64 rows x 128 cols, K-step 32.
// Wave (wr=w&3, wc=w>>2) owns a 16x64 strip -> 4 f32 16x16 accumulators.
// A is f32 (device input) or bf16 (attention context); converted in LDS.
// global_prefetch warms GL2 one K-step ahead.
// ---------------------------------------------------------------------------
template <bool A_BF16, bool OUT_BF16>
__global__ __launch_bounds__(256) void gemm_xwT_kernel(
    const void* __restrict__ A_, const float* __restrict__ W,
    const float* __restrict__ bias, void* __restrict__ C_) {
  constexpr int K = D_MODEL;
  constexpr int M = D_MODEL;

  __shared__ alignas(32) bf16 As[64][40];   // 64 rows x 32 k (pad 8)
  __shared__ alignas(32) bf16 Ws[128][48];  // 128 W-rows x 32 k (pad 16)

  const int tid  = threadIdx.x;
  const int lane = tid & 31;
  const int wave = tid >> 5;
  const int wr = wave & 3;        // wave row tile (16 rows each)
  const int wc = wave >> 2;       // wave col tile (64 cols each)
  const int rowBlock = blockIdx.x * 64;
  const int colBlock = blockIdx.y * 128;

  const int half = lane >> 4;     // 0: lanes 0-15, 1: lanes 16-31
  const int lr   = lane & 15;
  const int c0 = 16 * half;       // A-frag contiguous chunk offsets
  const int c1 = 16 * half + 8;

  const int arow = tid >> 2;            // 0..63
  const int acol = (tid & 3) * 8;       // 0,8,16,24
  const int wrow = tid >> 1;            // 0..127
  const int wcol = (tid & 1) * 16;      // 0,16

  v8f acc[4] = {};

  for (int k0 = 0; k0 < K; k0 += 32) {
    __syncthreads();
    if constexpr (A_BF16) {
      const bf16* Ag = (const bf16*)A_ + (size_t)(rowBlock + arow) * K + k0 + acol;
      *(v8bf*)&As[arow][acol] = *(const v8bf*)Ag;
      if (k0 + 32 < K) __builtin_prefetch(Ag + 32, 0, 1);
    } else {
      const float* Ag = (const float*)A_ + (size_t)(rowBlock + arow) * K + k0 + acol;
#pragma unroll
      for (int i = 0; i < 8; ++i) As[arow][acol + i] = (bf16)Ag[i];
      if (k0 + 32 < K) __builtin_prefetch(Ag + 32, 0, 1);
    }
    {
      const float* Wg = W + (size_t)(colBlock + wrow) * K + k0 + wcol;
#pragma unroll
      for (int i = 0; i < 16; ++i) Ws[wrow][wcol + i] = (bf16)Wg[i];
      if (k0 + 32 < K) __builtin_prefetch(Wg + 32, 0, 1);
    }
    __syncthreads();

    // A-frag: rows 16*wr..+15, contiguous 16 k at 16*half
    v16bf a = cat8(*(const v8bf*)&As[16 * wr + lr][c0],
                   *(const v8bf*)&As[16 * wr + lr][c1]);
#pragma unroll
    for (int f = 0; f < 4; ++f) {
      // B-frag: lane holds W-row (output col), contiguous 16 k at 16*half
      v16bf b = *(const v16bf*)&Ws[64 * wc + 16 * f + lr][16 * half];
      acc[f] = wmma_bf16(a, b, acc[f]);
    }
  }

  // C/D layout: VGPR j -> row j + 8*half, col = lane&15 within 16-col tile
#pragma unroll
  for (int f = 0; f < 4; ++f) {
    const int col = colBlock + 64 * wc + 16 * f + lr;
    const float bv = bias[col];
#pragma unroll
    for (int j = 0; j < 8; ++j) {
      const int row = rowBlock + 16 * wr + j + 8 * half;
      const float v = acc[f][j] + bv;
      if constexpr (OUT_BF16)
        ((bf16*)C_)[(size_t)row * M + col] = (bf16)v;
      else
        ((float*)C_)[(size_t)row * M + col] = v;
    }
  }
}

// ---------------------------------------------------------------------------
// Vproj[4096,1024] -> Vt[1024,4096]  (Vt[h*64+d][b*T+t]) tiled transpose.
// ---------------------------------------------------------------------------
__global__ __launch_bounds__(256) void transpose_vt_kernel(
    const bf16* __restrict__ V, bf16* __restrict__ Vt) {
  __shared__ bf16 tile[32][33];
  const int bx = blockIdx.x;              // along N (4096/32)
  const int by = blockIdx.y;              // along D (1024/32)
  const int tx = threadIdx.x & 31;
  const int ty = threadIdx.x >> 5;        // 0..7
#pragma unroll
  for (int p = 0; p < 4; ++p) {
    const int r = ty + p * 8;
    tile[r][tx] = V[(size_t)(bx * 32 + r) * D_MODEL + by * 32 + tx];
  }
  __syncthreads();
#pragma unroll
  for (int p = 0; p < 4; ++p) {
    const int r = ty + p * 8;
    Vt[(size_t)(by * 32 + r) * NROWS + bx * 32 + tx] = tile[tx][r];
  }
}

// ---------------------------------------------------------------------------
// Flash attention: one block per (b, h, 64 q-rows); 4 waves x 16 q-rows.
// K/V 64x64 tiles are TDM-streamed into double-buffered LDS; tile kt+1 DMA
// overlaps all WMMA/softmax work on tile kt. Online softmax over key tiles.
// S = Q K^T (8 wmma), O += P V (8 wmma) per tile.
// ---------------------------------------------------------------------------
__global__ __launch_bounds__(128) void attention_kernel(
    const bf16* __restrict__ Q, const bf16* __restrict__ Kp,
    const bf16* __restrict__ Vt, bf16* __restrict__ Ctx) {
  constexpr int NT = TSEQ / 64;           // 32 key tiles

  const int qt   = blockIdx.x;            // 0..T/64-1
  const int h    = blockIdx.y;
  const int b    = blockIdx.z;
  const int lane = threadIdx.x & 31;
  const int wave = threadIdx.x >> 5;      // 0..3
  const int half = lane >> 4;
  const int lr   = lane & 15;
  const int c0 = 16 * half;
  const int c1 = 16 * half + 8;
  const int qbase = qt * 64 + wave * 16;

  __shared__ alignas(32) bf16 Kt[2][64][64];     // key tile   (rows x d)
  __shared__ alignas(32) bf16 Vtl[2][64][64];    // value tile (d x keys)
  __shared__ alignas(32) bf16 Pbuf[4][16][72];   // per-wave P (pad 8)

  const bf16* kbase = Kp + (size_t)(b * TSEQ) * D_MODEL + h * DK;
  const bf16* vbase = Vt + (size_t)(h * DK) * NROWS + b * TSEQ;

  // prologue: wave 0 issues TDM for tile 0 into buffer 0
  if (wave == 0) {
    tdm_load_tile64((unsigned)(uintptr_t)&Kt[0][0][0],  kbase, D_MODEL,
                    (unsigned)D_MODEL, (unsigned)NROWS);
    tdm_load_tile64((unsigned)(uintptr_t)&Vtl[0][0][0], vbase, NROWS,
                    (unsigned)NROWS, (unsigned)D_MODEL);
  }

  // Q A-frags (held in registers across the whole k-loop)
  const bf16* qrow = Q + (size_t)(b * TSEQ + qbase + lr) * D_MODEL + h * DK;
  const v16bf qa0 = cat8(*(const v8bf*)(qrow + c0),      *(const v8bf*)(qrow + c1));
  const v16bf qa1 = cat8(*(const v8bf*)(qrow + 32 + c0), *(const v8bf*)(qrow + 32 + c1));

  v8f O[4] = {};
  float mrow[8], lsum[8];
#pragma unroll
  for (int j = 0; j < 8; ++j) { mrow[j] = -3.0e38f; lsum[j] = 0.0f; }

  for (int kt = 0; kt < NT; ++kt) {
    const int buf = kt & 1;

    // all waves are done reading buffer buf^1 (from iteration kt-1)
    __syncthreads();
    if (wave == 0) {
      if (kt + 1 < NT) {
        const int nb = (kt + 1) & 1;
        tdm_load_tile64((unsigned)(uintptr_t)&Kt[nb][0][0],
                        kbase + (size_t)(kt + 1) * 64 * D_MODEL, D_MODEL,
                        (unsigned)D_MODEL, (unsigned)NROWS);
        tdm_load_tile64((unsigned)(uintptr_t)&Vtl[nb][0][0],
                        vbase + (size_t)(kt + 1) * 64, NROWS,
                        (unsigned)NROWS, (unsigned)D_MODEL);
        __builtin_amdgcn_s_wait_tensorcnt((short)2);  // tile kt landed
      } else {
        __builtin_amdgcn_s_wait_tensorcnt((short)0);
      }
    }
    __syncthreads();   // publish tile kt to all waves

    // ---- S = Q K^T over d=64 (2 k-steps x 4 col tiles) ----
    v8f S[4] = {};
#pragma unroll
    for (int f = 0; f < 4; ++f) {
      const int kr = 16 * f + lr;   // key row within tile (lane's output col)
      const v16bf kb0 = *(const v16bf*)&Kt[buf][kr][16 * half];
      const v16bf kb1 = *(const v16bf*)&Kt[buf][kr][32 + 16 * half];
      S[f] = wmma_bf16(qa0, kb0, S[f]);
      S[f] = wmma_bf16(qa1, kb1, S[f]);
    }

    // ---- online softmax statistics (rows split per lane-half) ----
    float mnew[8], alpha[8], rsum[8];
#pragma unroll
    for (int j = 0; j < 8; ++j) {
      float mt = S[0][j];
#pragma unroll
      for (int f = 1; f < 4; ++f) mt = fmaxf(mt, S[f][j]);
      mt *= 0.125f;  // 1/sqrt(Dk)
#pragma unroll
      for (int m = 1; m <= 8; m <<= 1) mt = fmaxf(mt, __shfl_xor(mt, m, 32));
      mnew[j]  = fmaxf(mrow[j], mt);
      alpha[j] = __expf(mrow[j] - mnew[j]);
      mrow[j]  = mnew[j];
      rsum[j]  = 0.0f;
    }

    // ---- P = exp(S/8 - m); spill to LDS in C-layout for A-layout reload ----
#pragma unroll
    for (int f = 0; f < 4; ++f) {
#pragma unroll
      for (int j = 0; j < 8; ++j) {
        const float p = __expf(S[f][j] * 0.125f - mnew[j]);
        rsum[j] += p;
        Pbuf[wave][j + 8 * half][16 * f + lr] = (bf16)p;
      }
    }
#pragma unroll
    for (int j = 0; j < 8; ++j) {
      float rs = rsum[j];
#pragma unroll
      for (int m = 1; m <= 8; m <<= 1) rs += __shfl_xor(rs, m, 32);
      lsum[j] = alpha[j] * lsum[j] + rs;
    }
#pragma unroll
    for (int f = 0; f < 4; ++f)
#pragma unroll
      for (int j = 0; j < 8; ++j) O[f][j] *= alpha[j];

    __syncthreads();   // Pbuf written (wave-crossing lane layout change)

    // ---- P A-frags (16x64 -> two 16x32 frags) ----
    const v16bf pa0 = cat8(*(const v8bf*)&Pbuf[wave][lr][c0],
                           *(const v8bf*)&Pbuf[wave][lr][c1]);
    const v16bf pa1 = cat8(*(const v8bf*)&Pbuf[wave][lr][32 + c0],
                           *(const v8bf*)&Pbuf[wave][lr][32 + c1]);

    // ---- O += P V; V B-frags are contiguous rows of the LDS V tile ----
#pragma unroll
    for (int f = 0; f < 4; ++f) {
      const int vr = 16 * f + lr;   // d row within tile (lane's output col)
      const v16bf vb0 = *(const v16bf*)&Vtl[buf][vr][16 * half];
      const v16bf vb1 = *(const v16bf*)&Vtl[buf][vr][32 + 16 * half];
      O[f] = wmma_bf16(pa0, vb0, O[f]);
      O[f] = wmma_bf16(pa1, vb1, O[f]);
    }
  }

  // ---- epilogue: normalize and store context (bf16) ----
#pragma unroll
  for (int f = 0; f < 4; ++f) {
#pragma unroll
    for (int j = 0; j < 8; ++j) {
      const int row = b * TSEQ + qbase + j + 8 * half;
      const int col = h * DK + 16 * f + lr;
      Ctx[(size_t)row * D_MODEL + col] = (bf16)(O[f][j] / lsum[j]);
    }
  }
}

// ---------------------------------------------------------------------------
extern "C" void kernel_launch(void* const* d_in, const int* in_sizes, int n_in,
                              void* d_out, int out_size, void* d_ws,
                              size_t ws_size, hipStream_t stream) {
  (void)in_sizes; (void)n_in; (void)out_size; (void)ws_size;

  const float* query = (const float*)d_in[0];
  const float* key   = (const float*)d_in[1];
  const float* value = (const float*)d_in[2];
  const float* Wq = (const float*)d_in[3];
  const float* bq = (const float*)d_in[4];
  const float* Wk = (const float*)d_in[5];
  const float* bk = (const float*)d_in[6];
  const float* Wv = (const float*)d_in[7];
  const float* bv = (const float*)d_in[8];
  const float* Wo = (const float*)d_in[9];
  const float* bo = (const float*)d_in[10];
  float* out = (float*)d_out;

  // workspace layout (bf16): Qp | Kp | Vp | Vt | Ctx  -> 5 x 8 MB = 40 MB
  const size_t MAT = (size_t)NROWS * D_MODEL;
  bf16* Qp  = (bf16*)d_ws;
  bf16* Kp  = Qp + MAT;
  bf16* Vp  = Kp + MAT;
  bf16* Vt  = Vp + MAT;
  bf16* Ctx = Vt + MAT;

  dim3 gGrid(NROWS / 64, D_MODEL / 128);           // 64 x 8
  dim3 tGrid(NROWS / 32, D_MODEL / 32);            // 128 x 32
  dim3 aGrid(TSEQ / 64, HEADS, BATCH);             // 32 x 16 x 2

  gemm_xwT_kernel<false, true><<<gGrid, 256, 0, stream>>>(query, Wq, bq, Qp);
  gemm_xwT_kernel<false, true><<<gGrid, 256, 0, stream>>>(key,   Wk, bk, Kp);
  gemm_xwT_kernel<false, true><<<gGrid, 256, 0, stream>>>(value, Wv, bv, Vp);
  transpose_vt_kernel<<<tGrid, 256, 0, stream>>>(Vp, Vt);
  attention_kernel<<<aGrid, 128, 0, stream>>>(Qp, Kp, Vt, Ctx);
  gemm_xwT_kernel<true, false><<<gGrid, 256, 0, stream>>>(Ctx, Wo, bo, out);
}
